// MultiHeadAttention_17927193494305
// MI455X (gfx1250) — matrix operand order repaired and checked by reading, not compiled, not run on hardware
//
#include <hip/hip_runtime.h>
#include <hip/hip_bf16.h>

// Problem constants (from the reference)
#define BB 4
#define SS 2048
#define DD 768
#define HH 8
#define DEPTH 96
#define MTOT (BB * SS)   // 8192

typedef __bf16 bf16;
typedef __attribute__((ext_vector_type(16))) __bf16 v16bf;
typedef __attribute__((ext_vector_type(8)))  float  v8f;
typedef __attribute__((ext_vector_type(4)))  unsigned int u32x4;
typedef __attribute__((ext_vector_type(8)))  int i32x8;
typedef __attribute__((ext_vector_type(4)))  int i32x4;

union FragU { v16bf v; unsigned int u[8]; };

// ---------------------------------------------------------------------------
// Fragment loaders for V_WMMA_F32_16X16X32_BF16 (wave32; this code's contiguous k order, same for A and B)
// A (16x32, MxK): lane<16 -> row M=lane, K 0..15; lane>=16 -> K 16..31
// B (32x16, KxN): lane<16 -> col N=lane, K 0..15; lane>=16 -> K 16..31
// Source row-major [row][k] with k contiguous: each VGPR = one aligned u32.
// ---------------------------------------------------------------------------
__device__ __forceinline__ v16bf load_frag_a(const bf16* p, int stride,
                                             int rowbase, int koff, int lane) {
  const int r = lane & 15, hf = lane >> 4;
  const bf16* row = p + (size_t)(rowbase + r) * stride + koff;
  FragU f;
#pragma unroll
  for (int i = 0; i < 4; ++i) {
    f.u[i]     = *(const unsigned int*)(row + hf * 16 + 2 * i);
    f.u[i + 4] = *(const unsigned int*)(row + 8 + hf * 16 + 2 * i);
  }
  return f.v;
}

__device__ __forceinline__ v16bf load_frag_b(const bf16* p, int stride,
                                             int nbase, int koff, int lane) {
  const int n = lane & 15, hf = lane >> 4;
  const bf16* row = p + (size_t)(nbase + n) * stride + koff + hf * 16;
  FragU f;
#pragma unroll
  for (int i = 0; i < 8; ++i) f.u[i] = *(const unsigned int*)(row + 2 * i);
  return f.v;
}

// ---------------------------------------------------------------------------
// TDM: 2D tile load Global(bf16) -> LDS via tensor_load_to_lds (6-arg form,
// clang-23 / therock-10.0 toolchain).
// D# group0: {count=1, lds_addr, global_addr[56:0], type=2}
// D# group1: data_size=1 (2B), tensor dims == tile dims (tile fully in-bounds),
//            row stride in elements; groups 2/3 and trailing group = 0 (2D).
// Issued by one wave (TDM ignores EXEC); tracked by TENSORcnt.
// ---------------------------------------------------------------------------
__device__ __forceinline__ void tdm_load_2d(unsigned lds_off, const bf16* gptr,
                                            unsigned tile_d0, unsigned tile_d1,
                                            unsigned stride_d0) {
  const unsigned long long ga = (unsigned long long)(uintptr_t)gptr;
  u32x4 g0;
  g0[0] = 1u;                                               // count=1
  g0[1] = lds_off;                                          // LDS byte addr
  g0[2] = (unsigned)(ga & 0xffffffffull);                   // global addr lo
  g0[3] = (unsigned)((ga >> 32) & 0x1ffffffull) | (2u << 30); // hi | type=2
  i32x8 g1;
  g1[0] = (int)(1u << 16);                 // wg_mask=0, data_size=1 (2 bytes)
  g1[1] = (int)((tile_d0 & 0xffffu) << 16); // tensor_dim0[15:0] @ bits 63:48
  g1[2] = (int)((tile_d1 & 0xffffu) << 16); // tensor_dim1[15:0] @ bits 95:80
  g1[3] = (int)((tile_d0 & 0xffffu) << 16); // tile_dim0 @ bits 127:112
  g1[4] = (int)(tile_d1 & 0xffffu);         // tile_dim1 @ bits 143:128
  g1[5] = (int)stride_d0;                   // tensor_dim0_stride[31:0]
  g1[6] = 0;
  g1[7] = 0;
  const i32x4 z4 = (i32x4)0;
  const i32x8 z8 = (i32x8)0;
  __builtin_amdgcn_tensor_load_to_lds(g0, g1, z4, z4, z8, 0);
}

// ---------------------------------------------------------------------------
// GEMM 1: Y = X(fp32)[8192,768] @ W(fp32)[768,768] + bias, Y -> bf16.
// vtrans==0: Y stored as [B,H,S,DEPTH]   (Q and K)
// vtrans==1: Y stored as [B,H,DEPTH,S]   (V transposed for attention B-frags)
// ---------------------------------------------------------------------------
__global__ __launch_bounds__(256) void gemm_qkv_kernel(
    const float* __restrict__ X, const float* __restrict__ W,
    const float* __restrict__ bias, bf16* __restrict__ out, int vtrans) {
  __shared__ bf16 lA[128 * 34];
  __shared__ bf16 lB[128 * 34];

  const int tid = threadIdx.x, lane = tid & 31, w = tid >> 5;
  const int wm = w >> 1, wn = w & 1;
  const int m0 = blockIdx.y * 128, n0 = blockIdx.x * 128;
  const int K = DD, N = DD;

  v8f acc[2][4];
#pragma unroll
  for (int mi = 0; mi < 2; ++mi)
#pragma unroll
    for (int ni = 0; ni < 4; ++ni) acc[mi][ni] = (v8f)0.0f;

  for (int k0 = 0; k0 < K; k0 += 32) {
#pragma unroll
    for (int i = 0; i < 4; ++i) {
      const int idx = tid + i * 256;
      const int r = idx >> 3, c4 = (idx & 7) << 2;
      const float4 v = *(const float4*)&X[(size_t)(m0 + r) * K + k0 + c4];
      bf16* d = &lA[r * 34 + c4];
      d[0] = (bf16)v.x; d[1] = (bf16)v.y; d[2] = (bf16)v.z; d[3] = (bf16)v.w;
      if (k0 + 32 < K)
        __builtin_prefetch(&X[(size_t)(m0 + r) * K + k0 + 32 + c4], 0, 3);
    }
#pragma unroll
    for (int i = 0; i < 4; ++i) {
      const int idx = tid + i * 256;
      const int kk = idx >> 5, n4 = (idx & 31) << 2;
      const float4 v = *(const float4*)&W[(size_t)(k0 + kk) * N + n0 + n4];
      lB[(n4 + 0) * 34 + kk] = (bf16)v.x;
      lB[(n4 + 1) * 34 + kk] = (bf16)v.y;
      lB[(n4 + 2) * 34 + kk] = (bf16)v.z;
      lB[(n4 + 3) * 34 + kk] = (bf16)v.w;
      if (k0 + 32 < K)
        __builtin_prefetch(&W[(size_t)(k0 + 32 + kk) * N + n0 + n4], 0, 3);
    }
    __syncthreads();

    v16bf af[2];
#pragma unroll
    for (int mi = 0; mi < 2; ++mi)
      af[mi] = load_frag_a(lA, 34, wm * 32 + mi * 16, 0, lane);
#pragma unroll
    for (int ni = 0; ni < 4; ++ni) {
      const v16bf bfr = load_frag_b(lB, 34, wn * 64 + ni * 16, 0, lane);
#pragma unroll
      for (int mi = 0; mi < 2; ++mi)
        acc[mi][ni] = __builtin_amdgcn_wmma_f32_16x16x32_bf16(
            false, af[mi], false, bfr, (short)0, acc[mi][ni], false, false);
    }
    __syncthreads();
  }

  const int r = lane & 15, hf = lane >> 4;
#pragma unroll
  for (int mi = 0; mi < 2; ++mi)
#pragma unroll
    for (int ni = 0; ni < 4; ++ni) {
      const int col = n0 + wn * 64 + ni * 16 + r;
      const float bv = bias[col];
      const int h = col / DEPTH, dd = col % DEPTH;
#pragma unroll
      for (int vr = 0; vr < 8; ++vr) {
        const int m = m0 + wm * 32 + mi * 16 + vr + 8 * hf;
        const int bb = m >> 11, sr = m & (SS - 1);
        const float val = acc[mi][ni][vr] + bv;
        size_t idx;
        if (vtrans)
          idx = ((size_t)((bb * HH + h) * DEPTH + dd)) * SS + sr;
        else
          idx = ((size_t)((bb * HH + h) * SS + sr)) * DEPTH + dd;
        out[idx] = (bf16)val;
      }
    }
}

// ---------------------------------------------------------------------------
// GEMM 2: out = Ctx(bf16)[8192,768] @ Wo(fp32) + bo -> fp32 [8192,768]
// ---------------------------------------------------------------------------
__global__ __launch_bounds__(256) void gemm_out_kernel(
    const bf16* __restrict__ Ctx, const float* __restrict__ W,
    const float* __restrict__ bias, float* __restrict__ out) {
  __shared__ bf16 lA[128 * 34];
  __shared__ bf16 lB[128 * 34];

  const int tid = threadIdx.x, lane = tid & 31, w = tid >> 5;
  const int wm = w >> 1, wn = w & 1;
  const int m0 = blockIdx.y * 128, n0 = blockIdx.x * 128;
  const int K = DD, N = DD;

  v8f acc[2][4];
#pragma unroll
  for (int mi = 0; mi < 2; ++mi)
#pragma unroll
    for (int ni = 0; ni < 4; ++ni) acc[mi][ni] = (v8f)0.0f;

  for (int k0 = 0; k0 < K; k0 += 32) {
#pragma unroll
    for (int i = 0; i < 8; ++i) {
      const int idx = tid + i * 256;
      const int r = idx >> 4, c2 = (idx & 15) << 1;
      *(unsigned int*)&lA[r * 34 + c2] =
          *(const unsigned int*)&Ctx[(size_t)(m0 + r) * K + k0 + c2];
    }
#pragma unroll
    for (int i = 0; i < 4; ++i) {
      const int idx = tid + i * 256;
      const int kk = idx >> 5, n4 = (idx & 31) << 2;
      const float4 v = *(const float4*)&W[(size_t)(k0 + kk) * N + n0 + n4];
      lB[(n4 + 0) * 34 + kk] = (bf16)v.x;
      lB[(n4 + 1) * 34 + kk] = (bf16)v.y;
      lB[(n4 + 2) * 34 + kk] = (bf16)v.z;
      lB[(n4 + 3) * 34 + kk] = (bf16)v.w;
    }
    __syncthreads();

    v16bf af[2];
#pragma unroll
    for (int mi = 0; mi < 2; ++mi)
      af[mi] = load_frag_a(lA, 34, wm * 32 + mi * 16, 0, lane);
#pragma unroll
    for (int ni = 0; ni < 4; ++ni) {
      const v16bf bfr = load_frag_b(lB, 34, wn * 64 + ni * 16, 0, lane);
#pragma unroll
      for (int mi = 0; mi < 2; ++mi)
        acc[mi][ni] = __builtin_amdgcn_wmma_f32_16x16x32_bf16(
            false, af[mi], false, bfr, (short)0, acc[mi][ni], false, false);
    }
    __syncthreads();
  }

  const int r = lane & 15, hf = lane >> 4;
#pragma unroll
  for (int mi = 0; mi < 2; ++mi)
#pragma unroll
    for (int ni = 0; ni < 4; ++ni) {
      const int col = n0 + wn * 64 + ni * 16 + r;
      const float bv = bias[col];
#pragma unroll
      for (int vr = 0; vr < 8; ++vr) {
        const int m = m0 + wm * 32 + mi * 16 + vr + 8 * hf;
        out[(size_t)m * N + col] = acc[mi][ni][vr] + bv;
      }
    }
}

// ---------------------------------------------------------------------------
// Flash attention with TDM double-buffered K/V tiles.
// Block = 256 thr (8 waves); wave w owns query rows q0+16w..+15.
// Wave 0 issues TENSOR_LOAD_TO_LDS for the next 64-key tile while all waves
// compute on the current one. Post-softmax mask term is the q-independent
// ctx += 1e9 * (mask . V), accumulated separately (exact).
// ---------------------------------------------------------------------------
__global__ __launch_bounds__(256) void attention_kernel(
    const bf16* __restrict__ Q, const bf16* __restrict__ Kt,
    const bf16* __restrict__ Vt, const float* __restrict__ mask,
    bf16* __restrict__ ctxOut) {
  __shared__ bf16 lK[2][64 * 96];     // [kpos][depth], unpadded (TDM dest)
  __shared__ bf16 lV[2][96 * 64];     // [depth][kpos], unpadded (TDM dest)
  __shared__ bf16 lP[8 * 16 * 66];    // per-wave P relayout pads
  __shared__ float lMask[64];
  __shared__ float lMV[96];

  const int tid = threadIdx.x, lane = tid & 31, w = tid >> 5;
  const int bb = blockIdx.z, h = blockIdx.y, q0 = blockIdx.x * 128;
  const int qrow = q0 + w * 16;

  const bf16* Qg = Q + ((size_t)(bb * HH + h) * SS) * DEPTH;
  const bf16* Kg = Kt + ((size_t)(bb * HH + h) * SS) * DEPTH;
  const bf16* Vg = Vt + ((size_t)(bb * HH + h) * DEPTH) * SS;
  const float* mg = mask + (size_t)bb * SS;

  v16bf qf[3];
#pragma unroll
  for (int ks = 0; ks < 3; ++ks)
    qf[ks] = load_frag_a(Qg, DEPTH, qrow, ks * 32, lane);

  v8f ctx[6];
#pragma unroll
  for (int ni = 0; ni < 6; ++ni) ctx[ni] = (v8f)0.0f;
  float mrun[8], srun[8];
#pragma unroll
  for (int vr = 0; vr < 8; ++vr) { mrun[vr] = -1e30f; srun[vr] = 0.0f; }
  float mv = 0.0f;

  const float sc = 0.102062072615966f;  // 1/sqrt(96)
  bf16* pw = &lP[w * 16 * 66];

  // preload tile 0 (wave 0 issues; TDM ignores EXEC, descriptors uniform)
  if (w == 0) {
    tdm_load_2d((unsigned)(uintptr_t)&lK[0][0], Kg, DEPTH, 64, DEPTH);
    tdm_load_2d((unsigned)(uintptr_t)&lV[0][0], Vg, 64, DEPTH, SS);
  }

  for (int kt = 0; kt < SS; kt += 64) {
    const int cur = (kt >> 6) & 1;
    if (tid < 64) lMask[tid] = mg[kt + tid];

    if (w == 0) __builtin_amdgcn_s_wait_tensorcnt(0);
    __syncthreads();  // barrier A: tile `cur` visible to all waves

    // kick off next tile into the other buffer (free since barrier B of i-1)
    if (w == 0 && kt + 64 < SS) {
      const int nxt = cur ^ 1;
      tdm_load_2d((unsigned)(uintptr_t)&lK[nxt][0],
                  Kg + (size_t)(kt + 64) * DEPTH, DEPTH, 64, DEPTH);
      tdm_load_2d((unsigned)(uintptr_t)&lV[nxt][0],
                  Vg + (kt + 64), 64, DEPTH, SS);
    }

    // ---- scores = Q K^T (16 x 64 per wave) ----
    v8f sacc[4];
#pragma unroll
    for (int ni = 0; ni < 4; ++ni) sacc[ni] = (v8f)0.0f;
#pragma unroll
    for (int ks = 0; ks < 3; ++ks) {
#pragma unroll
      for (int ni = 0; ni < 4; ++ni) {
        const v16bf bfr = load_frag_b(&lK[cur][0], 96, ni * 16, ks * 32, lane);
        sacc[ni] = __builtin_amdgcn_wmma_f32_16x16x32_bf16(
            false, qf[ks], false, bfr, (short)0, sacc[ni], false, false);
      }
    }

    // ---- online softmax: row max via 16-lane xor shuffles, rescale ----
    float nm[8], corr[8];
#pragma unroll
    for (int vr = 0; vr < 8; ++vr) {
      float v = -1e30f;
#pragma unroll
      for (int ni = 0; ni < 4; ++ni) {
        sacc[ni][vr] *= sc;
        v = fmaxf(v, sacc[ni][vr]);
      }
#pragma unroll
      for (int off = 1; off < 16; off <<= 1)
        v = fmaxf(v, __shfl_xor(v, off, 32));
      nm[vr] = fmaxf(mrun[vr], v);
      corr[vr] = __expf(mrun[vr] - nm[vr]);
      mrun[vr] = nm[vr];
      srun[vr] *= corr[vr];
    }
#pragma unroll
    for (int ni = 0; ni < 6; ++ni)
#pragma unroll
      for (int vr = 0; vr < 8; ++vr) ctx[ni][vr] *= corr[vr];

    // ---- P = exp(s - m), row sums, C-frag -> A-frag via per-wave LDS ----
#pragma unroll
    for (int ni = 0; ni < 4; ++ni)
#pragma unroll
      for (int vr = 0; vr < 8; ++vr)
        sacc[ni][vr] = __expf(sacc[ni][vr] - nm[vr]);
#pragma unroll
    for (int vr = 0; vr < 8; ++vr) {
      float ps = sacc[0][vr] + sacc[1][vr] + sacc[2][vr] + sacc[3][vr];
#pragma unroll
      for (int off = 1; off < 16; off <<= 1) ps += __shfl_xor(ps, off, 32);
      srun[vr] += ps;
    }
    {
      const int r = lane & 15, hf = lane >> 4;
#pragma unroll
      for (int ni = 0; ni < 4; ++ni)
#pragma unroll
        for (int vr = 0; vr < 8; ++vr)
          pw[(vr + 8 * hf) * 66 + ni * 16 + r] = (bf16)sacc[ni][vr];
    }

    // ---- ctx += P @ V ----
#pragma unroll
    for (int kp = 0; kp < 2; ++kp) {
      const v16bf ap = load_frag_a(pw, 66, 0, kp * 32, lane);
#pragma unroll
      for (int ni = 0; ni < 6; ++ni) {
        const v16bf bv = load_frag_b(&lV[cur][0], 64, ni * 16, kp * 32, lane);
        ctx[ni] = __builtin_amdgcn_wmma_f32_16x16x32_bf16(
            false, ap, false, bv, (short)0, ctx[ni], false, false);
      }
    }

    // ---- post-softmax mask term: mv[d] += sum_k mask[k]*V[k,d] ----
    if (tid < DEPTH) {
      float a = 0.0f;
#pragma unroll
      for (int j = 0; j < 64; ++j) a += lMask[j] * (float)lV[cur][tid * 64 + j];
      mv += a;
    }
    __syncthreads();  // barrier B: done reading tile `cur`
  }

  if (tid < DEPTH) lMV[tid] = mv;
  __syncthreads();

  float rinv[8];
#pragma unroll
  for (int vr = 0; vr < 8; ++vr) rinv[vr] = 1.0f / srun[vr];
  const int r = lane & 15, hf = lane >> 4;
#pragma unroll
  for (int ni = 0; ni < 6; ++ni) {
    const int d = ni * 16 + r;
    const float madd = 1.0e9f * lMV[d];
    const int col = h * DEPTH + d;
#pragma unroll
    for (int vr = 0; vr < 8; ++vr) {
      const int srow = q0 + w * 16 + vr + 8 * hf;
      const float val = ctx[ni][vr] * rinv[vr] + madd;
      ctxOut[((size_t)(bb * SS + srow)) * DD + col] = (bf16)val;
    }
  }
}

// ---------------------------------------------------------------------------
extern "C" void kernel_launch(void* const* d_in, const int* in_sizes, int n_in,
                              void* d_out, int out_size, void* d_ws,
                              size_t ws_size, hipStream_t stream) {
  const float* query = (const float*)d_in[0];
  const float* key   = (const float*)d_in[1];
  const float* value = (const float*)d_in[2];
  const float* mask  = (const float*)d_in[3];
  const float* wq = (const float*)d_in[4];
  const float* bq = (const float*)d_in[5];
  const float* wk = (const float*)d_in[6];
  const float* bk = (const float*)d_in[7];
  const float* wv = (const float*)d_in[8];
  const float* bv = (const float*)d_in[9];
  const float* wo = (const float*)d_in[10];
  const float* bo = (const float*)d_in[11];
  float* out = (float*)d_out;

  // workspace layout (bf16): Q | K | Vt | Ctx, each B*H*S*DEPTH elems
  const size_t nElem = (size_t)BB * HH * SS * DEPTH;  // 6,291,456
  bf16* Qw = (bf16*)d_ws;
  bf16* Kw = Qw + nElem;
  bf16* Vw = Kw + nElem;
  bf16* Cw = Vw + nElem;   // ~50.3 MB of d_ws total

  dim3 blk(256);
  dim3 gg(DD / 128, MTOT / 128);  // (6, 64)
  gemm_qkv_kernel<<<gg, blk, 0, stream>>>(query, wq, bq, Qw, 0);
  gemm_qkv_kernel<<<gg, blk, 0, stream>>>(key,   wk, bk, Kw, 0);
  gemm_qkv_kernel<<<gg, blk, 0, stream>>>(value, wv, bv, Vw, 1);

  dim3 ga(SS / 128, HH, BB);      // (16, 8, 4)
  attention_kernel<<<ga, blk, 0, stream>>>(Qw, Kw, Vw, mask, Cw);

  gemm_out_kernel<<<gg, blk, 0, stream>>>(Cw, wo, bo, out);
}
